// S4ConditionalSideChainModel_14293651161726
// MI455X (gfx1250) — matrix operand order repaired and checked by reading, not compiled, not run on hardware
//
#include <hip/hip_runtime.h>
#include <hip/hip_bf16.h>
#include <stdint.h>

// ---------------------------------------------------------------------------
// S4D conditional side-chain model on gfx1250 (MI455X), FFT-free:
//   k1: conditioning MLP + per-h S4D tables (Toeplitz kernel M, boundary
//       weights Wv, state->output proj G, lambda^32), all f16 for WMMA.
//   k2: h = gelu(x @ W_lin + b_lin) via v_wmma_f32_16x16x32_f16, stored
//       transposed as u[B,H,L] in f16 (halves bandwidth).
//   k3: chunked linear scan (Mamba-2 style): per (b,h) block,
//       phase1: chunk summaries v[c] via WMMA -> 128KB LDS (CDNA5 big LDS),
//       phase2: sequential lambda^32 scan over 1024 chunks,
//       phase3: TDM tensor_load_to_lds stages u windows; intra-chunk conv +
//       state projection via WMMA (4 wmma/window), + D*u, write y[B,H,L] f32.
//   k4: LDS-transpose FiLM + gate: out = x * gelu(y*gamma + beta).
// Memory-bound (~0.75 GB total traffic @ 23.3 TB/s ~= 32us); WMMA carries all
// matrix math. gelu is branch-free (v_exp_f32 + v_rcp_f32) to keep VALU lean.
// ---------------------------------------------------------------------------

typedef __attribute__((ext_vector_type(16))) _Float16 v16h;
typedef __attribute__((ext_vector_type(8)))  _Float16 v8h;
typedef __attribute__((ext_vector_type(8)))  float    v8f;
typedef __attribute__((ext_vector_type(4)))  unsigned int u32x4;
typedef __attribute__((ext_vector_type(8)))  unsigned int u32x8;

#define B_  16
#define L_  32768
#define H_  64
#define N_  16
#define DM_ 32
#define T_  32            // chunk length
#define NC_ (L_ / T_)     // 1024 chunks per sequence
#define NW_ (NC_ / 16)    // 64 windows of 16 chunks

// workspace layout (bytes)
#define OFF_GB  0u                                  // gamma/beta f32 [B][2H]
#define OFF_M   8192u                               // M_h f16 [H][32][32]
#define OFF_WV  139264u                             // Wv f16 [H][2][16][32]
#define OFF_G   270336u                             // G f16 [H][32][32]
#define OFF_L32 401408u                             // lambda^32 f32 [H][16][2]
#define OFF_U   409600u                             // u f16 [B][H][L]
#define OFF_Y   (OFF_U + (unsigned)B_ * H_ * L_ * 2u) // y f32 [B][H][L]

// branch-free tanh: sign(x)*(1 - 2/(e^{2|x|}+1)); e->inf saturates to 1.
__device__ __forceinline__ float fast_tanh(float x) {
  float ax = fabsf(x);
  float e = __expf(2.0f * ax);                       // v_exp_f32 path
  float t = 1.0f - 2.0f * __builtin_amdgcn_rcpf(e + 1.0f);
  return copysignf(t, x);
}

__device__ __forceinline__ float gelu_f(float x) {
  float x3 = x * x * x;
  return 0.5f * x * (1.0f + fast_tanh(0.7978845608028654f * (x + 0.044715f * x3)));
}

// Load one 16x32 f16 A-matrix row per lane, contiguous K order:
// lanes 0-15: K 0..15 in halves 0..15 (row r)
// lanes 16-31: K 16..31 (same row r)
__device__ __forceinline__ v16h load_A(const _Float16* __restrict__ row, int jsel) {
  v8h lo = *(const v8h*)(row + 2 * jsel);
  v8h hi = *(const v8h*)(row + 2 * jsel + 8);
  v16h a;
#pragma unroll
  for (int i = 0; i < 8; ++i) { a[i] = lo[i]; a[8 + i] = hi[i]; }
  return a;
}

// ------------------------------- kernel 1 ----------------------------------
__global__ void k_precompute(const float* __restrict__ cp,
                             const float* __restrict__ W0, const float* __restrict__ b0,
                             const float* __restrict__ W1, const float* __restrict__ b1,
                             const float* __restrict__ W2, const float* __restrict__ b2,
                             const float* __restrict__ Wf, const float* __restrict__ bf,
                             const float* __restrict__ logdt,
                             const float* __restrict__ Are, const float* __restrict__ Aim,
                             const float* __restrict__ Cre, const float* __restrict__ Cim,
                             char* __restrict__ ws) {
  int t = threadIdx.x;
  float*     gb  = (float*)(ws + OFF_GB);
  _Float16*  M   = (_Float16*)(ws + OFF_M);
  _Float16*  Wv  = (_Float16*)(ws + OFF_WV);
  _Float16*  G   = (_Float16*)(ws + OFF_G);
  float*     l32 = (float*)(ws + OFF_L32);

  if (t < H_) {                       // per-head S4D tables
    int h = t;
    float dt = __expf(logdt[h]);
    float lre[N_], lim[N_], ctre[N_], ctim[N_];
    for (int n = 0; n < N_; ++n) {
      float ar = Are[h * N_ + n], ai = Aim[h * N_ + n];
      float er = __expf(dt * ar);
      float lr = er * __cosf(dt * ai), li = er * __sinf(dt * ai);
      lre[n] = lr; lim[n] = li;
      float nr = lr - 1.0f, ni = li;               // lambda - 1
      float den = ar * ar + ai * ai;
      float qr = (nr * ar + ni * ai) / den;        // (lambda-1)/A
      float qi = (ni * ar - nr * ai) / den;
      float cr = Cre[h * N_ + n], ci = Cim[h * N_ + n];
      ctre[n] = cr * qr - ci * qi;
      ctim[n] = cr * qi + ci * qr;
    }
    float K[T_];
    for (int d = 0; d < T_; ++d) K[d] = 0.0f;
    for (int n = 0; n < N_; ++n) {
      float pr = 1.0f, pi = 0.0f;                  // lambda^d
      for (int d = 0; d < T_; ++d) {
        K[d] += 2.0f * (ctre[n] * pr - ctim[n] * pi);
        Wv[(h * 2 + 0) * 512 + n * 32 + (31 - d)] = (_Float16)pr;  // Re lambda^{31-j}
        Wv[(h * 2 + 1) * 512 + n * 32 + (31 - d)] = (_Float16)pi;  // Im
        float npr = pr * lre[n] - pi * lim[n];
        float npi = pr * lim[n] + pi * lre[n];
        pr = npr; pi = npi;                         // now lambda^{d+1}
        float ga = ctre[n] * pr - ctim[n] * pi;     // Re(Ct lambda^{l+1})
        float gi = ctre[n] * pi + ctim[n] * pr;     // Im
        G[h * 1024 + d * 32 + n]      = (_Float16)(2.0f * ga);
        G[h * 1024 + d * 32 + 16 + n] = (_Float16)(-2.0f * gi);
      }
      l32[(h * N_ + n) * 2 + 0] = pr;               // lambda^32
      l32[(h * N_ + n) * 2 + 1] = pi;
    }
    for (int l = 0; l < T_; ++l)
      for (int j = 0; j < T_; ++j)
        M[h * 1024 + l * 32 + j] = (l >= j) ? (_Float16)K[l - j] : (_Float16)0.0f;
  }

  if (t >= 64 && t < 64 + B_) {       // conditioning MLP -> FiLM params
    int b = t - 64;
    float h0[DM_], h1[DM_];
    float x0 = cp[b * 2 + 0], x1 = cp[b * 2 + 1];
    for (int d = 0; d < DM_; ++d) h0[d] = gelu_f(x0 * W0[d] + x1 * W0[DM_ + d] + b0[d]);
    for (int d = 0; d < DM_; ++d) {
      float s = b1[d];
      for (int k = 0; k < DM_; ++k) s += h0[k] * W1[k * DM_ + d];
      h1[d] = gelu_f(s);
    }
    for (int d = 0; d < DM_; ++d) {
      float s = b2[d];
      for (int k = 0; k < DM_; ++k) s += h1[k] * W2[k * DM_ + d];
      h0[d] = gelu_f(s);
    }
    for (int o = 0; o < 2 * H_; ++o) {
      float s = bf[o];
      for (int k = 0; k < DM_; ++k) s += h0[k] * Wf[k * 2 * H_ + o];
      gb[b * 2 * H_ + o] = s;
    }
  }
}

// ------------------------------- kernel 2 ----------------------------------
// u[b,h,l] = gelu(x[b,l,:] @ W_lin[:,h] + b_lin[h]), 16x16 tile per wave.
__global__ void k_ingemm(const float* __restrict__ x, const float* __restrict__ Wlin,
                         const float* __restrict__ blin, char* __restrict__ ws) {
  __shared__ _Float16 tile[8][16][18];
  _Float16* u = (_Float16*)(ws + OFF_U);
  int lane = threadIdx.x & 31;
  int w = threadIdx.x >> 5;
  int g = blockIdx.x * 8 + w;
  int tile_m = g >> 2, tile_n = g & 3;
  int row0 = tile_m * 16;             // flat row in [B*L]
  int h0 = tile_n * 16;
  int b = row0 >> 15;                 // /L
  int l0 = row0 & (L_ - 1);

  int r = lane & 15;
  const float* xr = x + (size_t)(row0 + r) * H_;
  int k0 = (lane < 16) ? 0 : 8;
  v16h a0, a1;
#pragma unroll
  for (int i = 0; i < 8; ++i) {
    a0[i]     = (_Float16)xr[k0 + i];
    a0[8 + i] = (_Float16)xr[k0 + 16 + i];
    a1[i]     = (_Float16)xr[k0 + 32 + i];
    a1[8 + i] = (_Float16)xr[k0 + 48 + i];
  }
  int col = h0 + r;
  int kb = (lane < 16) ? 0 : 8;
  v16h bm0, bm1;
#pragma unroll
  for (int i = 0; i < 16; ++i) {
    bm0[i] = (_Float16)Wlin[(kb + (i & 7) + ((i >> 3) << 4)) * H_ + col];
    bm1[i] = (_Float16)Wlin[(kb + 32 + (i & 7) + ((i >> 3) << 4)) * H_ + col];
  }
  v8f c = {};
  c = __builtin_amdgcn_wmma_f32_16x16x32_f16(false, a0, false, bm0, (short)0, c, false, false);
  c = __builtin_amdgcn_wmma_f32_16x16x32_f16(false, a1, false, bm1, (short)0, c, false, false);

  float bias = blin[col];
  int mo = (lane < 16) ? 0 : 8;
#pragma unroll
  for (int i = 0; i < 8; ++i) tile[w][mo + i][r] = (_Float16)gelu_f(c[i] + bias);
  __syncthreads();
  // transpose out of D layout: 2 lanes per output h row, 8 f16 (16B) each
  int hl = lane >> 1, seg = lane & 1;
  v8h ov;
#pragma unroll
  for (int i = 0; i < 8; ++i) ov[i] = tile[w][seg * 8 + i][hl];
  _Float16* dst = u + ((size_t)(b * H_ + h0 + hl) << 15) + l0 + seg * 8;
  *(v8h*)dst = ov;
}

// ------------------------------- kernel 3 ----------------------------------
// One block per (b,h). Dynamic LDS: float2 sV[1024][16] (128KB) + 8x1KB stage.
__global__ void k_scan(char* __restrict__ ws, const float* __restrict__ Dvec) {
  extern __shared__ char smem[];
  float2* sV = (float2*)smem;
  const int STAGE0 = NC_ * N_ * (int)sizeof(float2);   // 131072
  int lane = threadIdx.x & 31;
  int wid = threadIdx.x >> 5;
  int bh = blockIdx.x;
  int h = bh & (H_ - 1);
  const _Float16* u_bh = (const _Float16*)(ws + OFF_U) + (size_t)bh * L_;
  float*          y_bh = (float*)(ws + OFF_Y) + (size_t)bh * L_;
  const _Float16* Mb  = (const _Float16*)(ws + OFF_M)  + h * 1024;
  const _Float16* Wvb = (const _Float16*)(ws + OFF_WV) + h * 1024;
  const _Float16* Gb  = (const _Float16*)(ws + OFF_G)  + h * 1024;
  const float2*   l32 = (const float2*)(ws + OFF_L32);

  int r = lane & 15;
  int jsel = (lane < 16) ? 0 : 8;
  v16h aWr = load_A(Wvb + r * 32, jsel);
  v16h aWi = load_A(Wvb + 512 + r * 32, jsel);

  // phase 1: chunk summaries v[c] = sum_j lambda^{31-j} u_j via WMMA
  for (int w = wid; w < NW_; w += 8) {
    int c0 = w * 16;
    v16h U = *(const v16h*)(u_bh + (size_t)(c0 + r) * 32 + ((lane < 16) ? 0 : 16));
    v8f z = {};
    v8f vre = __builtin_amdgcn_wmma_f32_16x16x32_f16(false, aWr, false, U, (short)0, z, false, false);
    v8f vim = __builtin_amdgcn_wmma_f32_16x16x32_f16(false, aWi, false, U, (short)0, z, false, false);
    int mo = (lane < 16) ? 0 : 8;
    int cc = c0 + r;
#pragma unroll
    for (int i = 0; i < 8; ++i) sV[cc * 16 + mo + i] = make_float2(vre[i], vim[i]);
  }
  __syncthreads();

  // phase 2: sequential lambda^32 scan across 1024 chunks (lane n = state n)
  if (threadIdx.x < 16) {
    int n = threadIdx.x;
    float2 lam = l32[h * N_ + n];
    float sr = 0.0f, si = 0.0f;
    for (int c = 0; c < NC_; ++c) {
      float2 v = sV[c * 16 + n];
      sV[c * 16 + n] = make_float2(sr, si);        // chunk-start state
      float nsr = lam.x * sr - lam.y * si + v.x;
      float nsi = lam.x * si + lam.y * sr + v.y;
      sr = nsr; si = nsi;
    }
  }
  __syncthreads();

  // phase 3: intra-chunk conv + state projection, y = conv + D*u
  v16h aM0 = load_A(Mb + r * 32, jsel);
  v16h aM1 = load_A(Mb + (16 + r) * 32, jsel);
  v16h aG0 = load_A(Gb + r * 32, jsel);
  v16h aG1 = load_A(Gb + (16 + r) * 32, jsel);
  float Dh = Dvec[h];
  _Float16* stage = (_Float16*)(smem + STAGE0 + wid * 1024);
  unsigned lds_off = (unsigned)__builtin_amdgcn_readfirstlane(STAGE0 + wid * 1024);

  for (int w = wid; w < NW_; w += 8) {
    int c0 = w * 16;
    {   // TDM: stage the contiguous 1KB u-window into this wave's LDS slot
      unsigned long long ga = (unsigned long long)(uintptr_t)(u_bh + (size_t)c0 * 32);
      unsigned galo = (unsigned)__builtin_amdgcn_readfirstlane((int)(unsigned)(ga & 0xffffffffull));
      unsigned gahi = (unsigned)__builtin_amdgcn_readfirstlane(
          (int)(((unsigned)((ga >> 32) & 0x01ffffffull)) | (2u << 30))); // type=2
      u32x4 g0 = { 1u, lds_off, galo, gahi };                 // count=1, lds, gaddr
      u32x8 g1 = { (1u << 16),            // data_size=1 (2B elements)
                   (512u << 16),          // tensor_dim0 = 512
                   (1u << 16),            // tensor_dim1 = 1
                   (512u << 16),          // tile_dim0 = 512
                   0u,                    // tile_dim1/2 = 0
                   512u, 0u, 0u };        // tensor_dim0_stride = 512
      asm volatile("s_wait_dscnt 0x0" ::: "memory");
      asm volatile("tensor_load_to_lds %0, %1" :: "s"(g0), "s"(g1) : "memory");
      asm volatile("s_wait_tensorcnt 0x0" ::: "memory");
    }
    v16h U = *(const v16h*)(stage + (size_t)r * 32 + ((lane < 16) ? 0 : 16));
    v8f z = {};
    v8f Y0 = __builtin_amdgcn_wmma_f32_16x16x32_f16(false, aM0, false, U, (short)0, z, false, false);
    v8f Y1 = __builtin_amdgcn_wmma_f32_16x16x32_f16(false, aM1, false, U, (short)0, z, false, false);

    v16h S;   // B-matrix of chunk-start states: K<16 -> re, K>=16 -> im
    {
      const float2* p = sV + (c0 + r) * 16;
      if (lane < 16) {
#pragma unroll
        for (int i = 0; i < 16; ++i) S[i] = (_Float16)p[i].x;
      } else {
#pragma unroll
        for (int i = 0; i < 16; ++i) S[i] = (_Float16)p[i].y;
      }
    }
    Y0 = __builtin_amdgcn_wmma_f32_16x16x32_f16(false, aG0, false, S, (short)0, Y0, false, false);
    Y1 = __builtin_amdgcn_wmma_f32_16x16x32_f16(false, aG1, false, S, (short)0, Y1, false, false);

    int lo = (lane < 16) ? 0 : 8;
    int cc = c0 + r;
    float* yb = y_bh + (size_t)cc * T_;
    const _Float16* ub = stage + r * 32;
#pragma unroll
    for (int i = 0; i < 8; ++i) yb[lo + i] = Y0[i] + Dh * (float)ub[lo + i];
#pragma unroll
    for (int i = 0; i < 8; ++i) yb[16 + lo + i] = Y1[i] + Dh * (float)ub[16 + lo + i];
  }
}

// ------------------------------- kernel 4 ----------------------------------
// out[b,l,h] = x[b,l,h] * gelu(y[b,h,l]*gamma[b,h] + beta[b,h]); LDS transpose.
__global__ void k_film(const float* __restrict__ x, const char* __restrict__ ws,
                       float* __restrict__ out) {
  __shared__ float tile[64][65];
  const float* y  = (const float*)(ws + OFF_Y);
  const float* gb = (const float*)(ws + OFF_GB);
  int b = blockIdx.x >> 9;
  int lt = blockIdx.x & 511;
  int l0 = lt * 64;
  int t = threadIdx.x;
  {
    int h = t >> 2, l4 = (t & 3) * 16;
    const float* yr = y + ((size_t)(b * H_ + h) << 15) + l0 + l4;
#pragma unroll
    for (int i = 0; i < 16; ++i) tile[l4 + i][h] = yr[i];
  }
  __syncthreads();
  {
    int l = t >> 2, h4 = (t & 3) * 16;
    size_t base = ((size_t)b * L_ + (size_t)(l0 + l)) * H_ + h4;
#pragma unroll
    for (int i = 0; i < 16; ++i) {
      int h = h4 + i;
      float g  = gb[b * 128 + h];
      float bt = gb[b * 128 + 64 + h];
      float yy = gelu_f(tile[l][h] * g + bt);
      out[base + i] = x[base + i] * yy;
    }
  }
}

// ------------------------------- launcher ----------------------------------
extern "C" void kernel_launch(void* const* d_in, const int* in_sizes, int n_in,
                              void* d_out, int out_size, void* d_ws, size_t ws_size,
                              hipStream_t stream) {
  const float* x     = (const float*)d_in[0];
  const float* cp    = (const float*)d_in[1];
  const float* W0    = (const float*)d_in[2];
  const float* b0    = (const float*)d_in[3];
  const float* W1    = (const float*)d_in[4];
  const float* b1    = (const float*)d_in[5];
  const float* W2    = (const float*)d_in[6];
  const float* b2    = (const float*)d_in[7];
  const float* Wlin  = (const float*)d_in[8];
  const float* blin  = (const float*)d_in[9];
  const float* logdt = (const float*)d_in[10];
  const float* Are   = (const float*)d_in[11];
  const float* Aim   = (const float*)d_in[12];
  const float* Cre   = (const float*)d_in[13];
  const float* Cim   = (const float*)d_in[14];
  const float* Dv    = (const float*)d_in[15];
  const float* Wf    = (const float*)d_in[16];
  const float* bf    = (const float*)d_in[17];
  char* ws = (char*)d_ws;
  float* out = (float*)d_out;

  k_precompute<<<1, 256, 0, stream>>>(cp, W0, b0, W1, b1, W2, b2, Wf, bf,
                                      logdt, Are, Aim, Cre, Cim, ws);
  // (B*L/16 row-tiles) * 4 col-tiles, 8 waves (tiles) per 256-thread block
  k_ingemm<<<(B_ * (L_ / 16) * 4) / 8, 256, 0, stream>>>(x, Wlin, blin, ws);
  // one block per (b,h); 128KB chunk-state array + 8x1KB TDM staging in LDS
  k_scan<<<B_ * H_, 256, 131072 + 8 * 1024, stream>>>(ws, Dv);
  k_film<<<B_ * 512, 256, 0, stream>>>(x, ws, out);
}
